// Pair2Pair_188978561515
// MI455X (gfx1250) — matrix operand order repaired and checked by reading, not compiled, not run on hardware
//
#include <hip/hip_runtime.h>

// AF2 pair-stack block for MI455X (gfx1250, wave32, WMMA 16x16x32 bf16).
// All GEMM B-operands are stored transposed (N x K, K-contiguous) so both A and B
// fragments load as two 16-byte contiguous chunks per lane (global_load_b128 x2).
// Workspace requirement: ~103 MB (xln 16.8MB + 1MB weights + 85MB stage union).

typedef __bf16 bf16;
typedef __attribute__((ext_vector_type(16))) __bf16 v16bf;
typedef __attribute__((ext_vector_type(8)))  __bf16 v8bf;
typedef __attribute__((ext_vector_type(8)))  float  v8f;

#define LSEQ   256
#define DPAIR  128
#define NROWS  65536   // LSEQ*LSEQ
#define DHID   32
#define NHEAD  4

// Single v_rcp_f32 (1 ULP) instead of the precise-division expansion; result is
// bf16-rounded immediately, so the extra precision would be discarded anyway.
__device__ __forceinline__ float fast_rcp(float x) { return __builtin_amdgcn_rcpf(x); }
__device__ __forceinline__ float sigmoidf_(float x) { return fast_rcp(1.f + __expf(-x)); }

__device__ __forceinline__ v8f wmma_bf16(v16bf a, v16bf b, v8f c) {
  // D = A(16x32 bf16) * B(32x16 bf16) + C(16x16 f32)
  return __builtin_amdgcn_wmma_f32_16x16x32_bf16(false, a, false, b, (short)0, c, false, false);
}

__device__ __forceinline__ v16bf concat8(v8bf lo, v8bf hi) {
  return __builtin_shufflevector(lo, hi, 0, 1, 2, 3, 4, 5, 6, 7, 8, 9, 10, 11, 12, 13, 14, 15);
}

// A fragment: 16x32 bf16 tile, row-major (M x K), leading dim lda (multiple of 8).
// ISA layout: lane = M; lanes<16 hold K {0..7,16..23}, lanes>=16 hold {8..15,24..31}.
__device__ __forceinline__ v16bf load_a_frag(const bf16* A, int lda) {
  int lane = threadIdx.x & 31;
  const bf16* row = A + (size_t)(lane & 15) * lda + ((lane < 16) ? 0 : 8);
  v8bf lo = *(const v8bf*)(row);
  v8bf hi = *(const v8bf*)(row + 16);
  return concat8(lo, hi);
}

// B fragment from TRANSPOSED storage: Bt is (N x K) row-major, leading dim ldk.
// ISA layout: lane = N; lanes<16 hold K {0..7,16..23}, lanes>=16 hold {8..15,24..31}.
__device__ __forceinline__ v16bf load_bt_frag(const bf16* Bt, int ldk) {
  int lane = threadIdx.x & 31;
  const bf16* row = Bt + (size_t)(lane & 15) * ldk + ((lane < 16) ? 0 : 8);
  v8bf lo = *(const v8bf*)(row);
  v8bf hi = *(const v8bf*)(row + 16);
  return concat8(lo, hi);
}
// C fragment: lane holds col = lane&15, rows v + ((lane<16)?0:8), v=0..7.

// ---------------- weight cast: f32 (K x N) -> bf16 transposed (Ndst x K), zero-padded ----------------
__global__ void castT_kernel(const float* __restrict__ src, bf16* __restrict__ dst,
                             int K, int nsrc, int ndst) {
  int idx = blockIdx.x * blockDim.x + threadIdx.x;
  if (idx >= ndst * K) return;
  int n = idx / K, k = idx - n * K;
  float v = (n < nsrc) ? src[(size_t)k * nsrc + n] : 0.f;
  dst[idx] = (bf16)v;
}

// ---------------- LayerNorm over 128 (one wave per row) ----------------
__global__ void ln128_kernel(const float* __restrict__ x, const float* __restrict__ g,
                             const float* __restrict__ b, bf16* __restrict__ y, int transpose) {
  int row = (blockIdx.x * blockDim.x + threadIdx.x) >> 5;
  int lane = threadIdx.x & 31;
  if (row >= NROWS) return;
  int sr = transpose ? (((row & 255) << 8) | (row >> 8)) : row;
  const float* src = x + (size_t)sr * DPAIR;
  float v[4]; float s = 0.f;
#pragma unroll
  for (int t = 0; t < 4; ++t) { v[t] = src[lane + 32 * t]; s += v[t]; }
#pragma unroll
  for (int m = 16; m >= 1; m >>= 1) s += __shfl_xor(s, m, 32);
  float mean = s * (1.f / DPAIR);
  float var = 0.f;
#pragma unroll
  for (int t = 0; t < 4; ++t) { float d = v[t] - mean; var += d * d; }
#pragma unroll
  for (int m = 16; m >= 1; m >>= 1) var += __shfl_xor(var, m, 32);
  float rstd = rsqrtf(var * (1.f / DPAIR) + 1e-5f);
  bf16* dst = y + (size_t)row * DPAIR;
#pragma unroll
  for (int t = 0; t < 4; ++t) {
    int c = lane + 32 * t;
    dst[c] = (bf16)((v[t] - mean) * rstd * g[c] + b[c]);
  }
}

// ---------------- LayerNorm over 32 (one wave per row, one elem per lane) ----------------
__global__ void ln32_kernel(const float* __restrict__ x, const float* __restrict__ g,
                            const float* __restrict__ b, bf16* __restrict__ y) {
  int row = (blockIdx.x * blockDim.x + threadIdx.x) >> 5;
  int lane = threadIdx.x & 31;
  if (row >= NROWS) return;
  float v = x[(size_t)row * DHID + lane];
  float s = v;
#pragma unroll
  for (int m = 16; m >= 1; m >>= 1) s += __shfl_xor(s, m, 32);
  float mean = s * (1.f / DHID);
  float d = v - mean;
  float var = d * d;
#pragma unroll
  for (int m = 16; m >= 1; m >>= 1) var += __shfl_xor(var, m, 32);
  float rstd = rsqrtf(var * (1.f / DHID) + 1e-5f);
  y[(size_t)row * DHID + lane] = (bf16)(d * rstd * g[lane] + b[lane]);
}

// ---------------- tri-mul projections: left/right (gated) + pair gate ----------------
// grid (1024, 12), block 128 (4 waves). unit 0-1: left cols, 2-3: right cols, 4-11: gate cols.
// Lmat layout: [d][i][k] (A operand).  Rtm layout: [d][j][k] (B^T operand).
// outgoing: both are direct writes at pair row r; incoming: both transposed.
__global__ void trimul_proj_kernel(
    const bf16* __restrict__ xln,
    const bf16* lwT, const float* lb, const bf16* lgwT, const float* lgb,
    const bf16* rwT, const float* rb, const bf16* rgwT, const float* rgb,
    const bf16* gwT, const float* gb,
    bf16* __restrict__ Lmat, bf16* __restrict__ Rtm, bf16* __restrict__ Gate,
    int outgoing) {
  int wave = threadIdx.x >> 5;
  int lane = threadIdx.x & 31;
  int row0 = (blockIdx.x * 4 + wave) * 16;
  int unit = blockIdx.y;
  const bf16* Abase = xln + (size_t)row0 * DPAIR;
  int rowoff = (lane < 16) ? 0 : 8;

  if (unit < 4) {
    int isR = unit >> 1;
    int ct = (unit & 1) * 16;
    const bf16* vwt = isR ? rwT  : lwT;
    const bf16* gwt = isR ? rgwT : lgwT;
    const float* vb  = isR ? rb  : lb;
    const float* gb_ = isR ? rgb : lgb;
    v8f accV = {}; v8f accG = {};
#pragma unroll
    for (int ks = 0; ks < 4; ++ks) {
      v16bf a  = load_a_frag(Abase + ks * 32, DPAIR);
      v16bf bv = load_bt_frag(vwt + (size_t)ct * DPAIR + ks * 32, DPAIR);
      v16bf bg = load_bt_frag(gwt + (size_t)ct * DPAIR + ks * 32, DPAIR);
      accV = wmma_bf16(a, bv, accV);
      accG = wmma_bf16(a, bg, accG);
    }
    int col = ct + (lane & 15);
    float bV = vb[col], bG = gb_[col];
    bf16* dst = isR ? Rtm : Lmat;
#pragma unroll
    for (int v = 0; v < 8; ++v) {
      int r = row0 + rowoff + v;
      float act = (accV[v] + bV) * sigmoidf_(accG[v] + bG);
      size_t idx = (size_t)col * NROWS +
                   (outgoing ? (size_t)r : ((size_t)(r & 255) * LSEQ + (r >> 8)));
      dst[idx] = (bf16)act;
    }
  } else {
    int ct = (unit - 4) * 16;
    v8f acc = {};
#pragma unroll
    for (int ks = 0; ks < 4; ++ks) {
      v16bf a  = load_a_frag(Abase + ks * 32, DPAIR);
      v16bf bg = load_bt_frag(gwT + (size_t)ct * DPAIR + ks * 32, DPAIR);
      acc = wmma_bf16(a, bg, acc);
    }
    int col = ct + (lane & 15);
    float bG = gb[col];
#pragma unroll
    for (int v = 0; v < 8; ++v) {
      int r = row0 + rowoff + v;
      Gate[(size_t)r * DPAIR + col] = (bf16)sigmoidf_(acc[v] + bG);
    }
  }
}

// ---------------- triangle einsum: 32 batched 256x256x256 GEMMs, 32x32 block per wave ----------------
__global__ void tri_gemm_kernel(const bf16* __restrict__ Lmat, const bf16* __restrict__ Rtm,
                                float* __restrict__ Tri) {
  int gw = (blockIdx.x * blockDim.x + threadIdx.x) >> 5;  // 2048 waves
  int lane = threadIdx.x & 31;
  int d = gw >> 6;            // 64 32x32 blocks per channel
  int t = gw & 63;
  int i0 = (t >> 3) * 32, j0 = (t & 7) * 32;
  const bf16* A = Lmat + (size_t)d * NROWS + (size_t)i0 * LSEQ;
  const bf16* B = Rtm  + (size_t)d * NROWS + (size_t)j0 * LSEQ;
  v8f a00 = {}, a01 = {}, a10 = {}, a11 = {};
#pragma unroll
  for (int ks = 0; ks < 8; ++ks) {
    v16bf fa0 = load_a_frag(A + ks * 32, LSEQ);
    v16bf fa1 = load_a_frag(A + (size_t)16 * LSEQ + ks * 32, LSEQ);
    v16bf fb0 = load_bt_frag(B + ks * 32, LSEQ);
    v16bf fb1 = load_bt_frag(B + (size_t)16 * LSEQ + ks * 32, LSEQ);
    a00 = wmma_bf16(fa0, fb0, a00);
    a01 = wmma_bf16(fa0, fb1, a01);
    a10 = wmma_bf16(fa1, fb0, a10);
    a11 = wmma_bf16(fa1, fb1, a11);
  }
  int rowoff = (lane < 16) ? 0 : 8;
  int cl = lane & 15;
#pragma unroll
  for (int v = 0; v < 8; ++v) {
    int i = i0 + rowoff + v;
    Tri[((size_t)i * LSEQ + j0 + cl) * DHID + d]            = a00[v];
    Tri[((size_t)i * LSEQ + j0 + 16 + cl) * DHID + d]       = a01[v];
    Tri[((size_t)(i + 16) * LSEQ + j0 + cl) * DHID + d]     = a10[v];
    Tri[((size_t)(i + 16) * LSEQ + j0 + 16 + cl) * DHID + d] = a11[v];
  }
}

// ---------------- tri-mul output projection + gate + residual ----------------
__global__ void trimul_out_kernel(const bf16* __restrict__ TriLn, const bf16* __restrict__ owT,
                                  const float* __restrict__ ob, const bf16* __restrict__ Gate,
                                  const float* __restrict__ pair_in, float* __restrict__ pair_out) {
  int wave = threadIdx.x >> 5;
  int lane = threadIdx.x & 31;
  int row0 = (blockIdx.x * 4 + wave) * 16;
  int ct = blockIdx.y * 16;
  v16bf a = load_a_frag(TriLn + (size_t)row0 * DHID, DHID);       // K = 32, one step
  v16bf b = load_bt_frag(owT + (size_t)ct * DHID, DHID);
  v8f acc = {};
  acc = wmma_bf16(a, b, acc);
  int col = ct + (lane & 15);
  float bias = ob[col];
  int rowoff = (lane < 16) ? 0 : 8;
#pragma unroll
  for (int v = 0; v < 8; ++v) {
    size_t r = (size_t)row0 + rowoff + v;
    float gate = (float)Gate[r * DPAIR + col];
    pair_out[r * DPAIR + col] = pair_in[r * DPAIR + col] + gate * (acc[v] + bias);
  }
}

// ---------------- attention projections q/k/v/g/bias with layout scatter ----------------
// grid (1024, 33): unit 0-7 q, 8-15 k, 16-23 v, 24-31 g, 32 pair-bias.
// qmat: [i,h][j][d] (A op).  kmat: [i,h][k2][d] (B^T of QK).  vmat: [i,h][d][k2] (B^T of PV).
__global__ void attn_proj_kernel(
    const bf16* __restrict__ xln,
    const bf16* qwT, const bf16* kwT, const bf16* vwT, const bf16* gwT, const float* gb,
    const bf16* bwT,
    bf16* __restrict__ qmat, bf16* __restrict__ kmat, bf16* __restrict__ vmat,
    bf16* __restrict__ gbuf, float* __restrict__ bbuf) {
  int wave = threadIdx.x >> 5;
  int lane = threadIdx.x & 31;
  int row0 = (blockIdx.x * 4 + wave) * 16;
  int unit = blockIdx.y;
  int type = unit >> 3;
  int ct = (unit & 7) * 16;
  const bf16* Wt;
  if (unit == 32) { Wt = bwT; ct = 0; type = 4; }
  else { Wt = (type == 0) ? qwT : (type == 1) ? kwT : (type == 2) ? vwT : gwT; }
  const bf16* Abase = xln + (size_t)row0 * DPAIR;
  v8f acc = {};
#pragma unroll
  for (int ks = 0; ks < 4; ++ks) {
    v16bf a = load_a_frag(Abase + ks * 32, DPAIR);
    v16bf b = load_bt_frag(Wt + (size_t)ct * DPAIR + ks * 32, DPAIR);
    acc = wmma_bf16(a, b, acc);
  }
  int col = ct + (lane & 15);
  int rowoff = (lane < 16) ? 0 : 8;
#pragma unroll
  for (int v = 0; v < 8; ++v) {
    int r = row0 + rowoff + v;
    int i = r >> 8, jq = r & 255;
    int h = col >> 5, dd = col & 31;
    if (type == 0) {        // q, pre-scaled by 1/sqrt(32)
      qmat[(((size_t)i * NHEAD + h) * LSEQ + jq) * DHID + dd] = (bf16)(acc[v] * 0.17677669529663687f);
    } else if (type == 1) { // k: [i,h][k2][d]
      kmat[(((size_t)i * NHEAD + h) * LSEQ + jq) * DHID + dd] = (bf16)acc[v];
    } else if (type == 2) { // v: [i,h][d][k2]
      vmat[(((size_t)i * NHEAD + h) * DHID + dd) * LSEQ + jq] = (bf16)acc[v];
    } else if (type == 3) { // gate
      gbuf[(size_t)r * DPAIR + col] = (bf16)sigmoidf_(acc[v] + gb[col]);
    } else {                // pair bias: bbuf[h][j][k]
      if (col < NHEAD) bbuf[(size_t)col * NROWS + r] = acc[v];
    }
  }
}

// ---------------- flash-style triangle attention core ----------------
// grid (1024 = i*4+h, 4), block 128. Wave handles 16 query rows, k-blocks of 32.
__global__ void attn_flash_kernel(
    const bf16* __restrict__ qmat, const bf16* __restrict__ kmat, const bf16* __restrict__ vmat,
    const float* __restrict__ bbuf, const bf16* __restrict__ gbuf, bf16* __restrict__ obuf) {
  __shared__ __align__(16) bf16 plds[4][16 * 32];
  int wave = threadIdx.x >> 5;
  int lane = threadIdx.x & 31;
  int ih = blockIdx.x;
  int i = ih >> 2, h = ih & 3;
  int jt = blockIdx.y * 4 + wave;
  int rowoff = (lane < 16) ? 0 : 8;
  int colL = lane & 15;

  v16bf aq = load_a_frag(qmat + ((size_t)ih * LSEQ + jt * 16) * DHID, DHID);
  v8f o0 = {}, o1 = {};
  float mrow[8], lrow[8];
#pragma unroll
  for (int v = 0; v < 8; ++v) { mrow[v] = -1e30f; lrow[v] = 0.f; }
  const float* biasrow = bbuf + (size_t)h * NROWS;
  const bf16* Kbase = kmat + (size_t)ih * LSEQ * DHID;   // [k2][d], ld 32
  const bf16* Vbase = vmat + (size_t)ih * DHID * LSEQ;   // [d][k2], ld 256
  bf16* pl = plds[wave];

  for (int kb = 0; kb < 8; ++kb) {
    v16bf b0 = load_bt_frag(Kbase + (size_t)(kb * 32) * DHID, DHID);
    v16bf b1 = load_bt_frag(Kbase + (size_t)(kb * 32 + 16) * DHID, DHID);
    v8f zero = {};
    v8f s0 = wmma_bf16(aq, b0, zero);
    v8f s1 = wmma_bf16(aq, b1, zero);
    float alpha[8];
#pragma unroll
    for (int v = 0; v < 8; ++v) {
      int j = jt * 16 + rowoff + v;
      float x0 = s0[v] + biasrow[(size_t)j * LSEQ + kb * 32 + colL];
      float x1 = s1[v] + biasrow[(size_t)j * LSEQ + kb * 32 + 16 + colL];
      float mx = fmaxf(x0, x1);
#pragma unroll
      for (int m = 8; m >= 1; m >>= 1) mx = fmaxf(mx, __shfl_xor(mx, m, 32));
      float mn = fmaxf(mrow[v], mx);
      alpha[v] = __expf(mrow[v] - mn);
      mrow[v] = mn;
      float p0 = __expf(x0 - mn), p1 = __expf(x1 - mn);
      float ps = p0 + p1;
#pragma unroll
      for (int m = 8; m >= 1; m >>= 1) ps += __shfl_xor(ps, m, 32);
      lrow[v] = lrow[v] * alpha[v] + ps;
      int rr = rowoff + v;
      pl[rr * 32 + colL] = (bf16)p0;
      pl[rr * 32 + 16 + colL] = (bf16)p1;
    }
    asm volatile("s_wait_dscnt 0" ::: "memory");
    __builtin_amdgcn_wave_barrier();
    v16bf ap  = load_a_frag(pl, 32);                       // C-frag -> A-frag via LDS
    v16bf bv0 = load_bt_frag(Vbase + kb * 32, LSEQ);
    v16bf bv1 = load_bt_frag(Vbase + (size_t)16 * LSEQ + kb * 32, LSEQ);
#pragma unroll
    for (int v = 0; v < 8; ++v) { o0[v] *= alpha[v]; o1[v] *= alpha[v]; }
    o0 = wmma_bf16(ap, bv0, o0);
    o1 = wmma_bf16(ap, bv1, o1);
    asm volatile("s_wait_dscnt 0" ::: "memory");
    __builtin_amdgcn_wave_barrier();
  }
#pragma unroll
  for (int v = 0; v < 8; ++v) {
    int j = jt * 16 + rowoff + v;
    size_t r = (size_t)i * LSEQ + j;
    float inv = fast_rcp(lrow[v]);
    float g0 = (float)gbuf[r * DPAIR + h * DHID + colL];
    float g1 = (float)gbuf[r * DPAIR + h * DHID + 16 + colL];
    obuf[r * DPAIR + h * DHID + colL]      = (bf16)(g0 * o0[v] * inv);
    obuf[r * DPAIR + h * DHID + 16 + colL] = (bf16)(g1 * o1[v] * inv);
  }
}

// ---------------- FF first GEMM with ReLU ----------------
__global__ void ff1_kernel(const bf16* __restrict__ xln, const bf16* __restrict__ w1T,
                           const float* __restrict__ b1, bf16* __restrict__ hbuf) {
  int wave = threadIdx.x >> 5;
  int lane = threadIdx.x & 31;
  int row0 = (blockIdx.x * 4 + wave) * 16;
  int ct = blockIdx.y * 16;
  const bf16* Abase = xln + (size_t)row0 * DPAIR;
  v8f acc = {};
#pragma unroll
  for (int ks = 0; ks < 4; ++ks) {
    v16bf a = load_a_frag(Abase + ks * 32, DPAIR);
    v16bf b = load_bt_frag(w1T + (size_t)ct * DPAIR + ks * 32, DPAIR);
    acc = wmma_bf16(a, b, acc);
  }
  int col = ct + (lane & 15);
  float bias = b1[col];
  int rowoff = (lane < 16) ? 0 : 8;
#pragma unroll
  for (int v = 0; v < 8; ++v) {
    size_t r = (size_t)row0 + rowoff + v;
    hbuf[r * 256 + col] = (bf16)fmaxf(acc[v] + bias, 0.f);
  }
}

// ---------------- generic GEMM (N=128) + bias + residual (optional transpose) ----------------
__global__ void gemm_resid_kernel(const bf16* __restrict__ A, int K,
                                  const bf16* __restrict__ Wt, const float* __restrict__ bias,
                                  const float* __restrict__ pair_in, float* __restrict__ pair_out,
                                  int transpose) {
  int wave = threadIdx.x >> 5;
  int lane = threadIdx.x & 31;
  int row0 = (blockIdx.x * 4 + wave) * 16;
  int ct = blockIdx.y * 16;
  const bf16* Abase = A + (size_t)row0 * K;
  const bf16* Wbase = Wt + (size_t)ct * K;
  v8f acc = {};
  for (int ks = 0; ks < K / 32; ++ks) {
    v16bf a = load_a_frag(Abase + ks * 32, K);
    v16bf b = load_bt_frag(Wbase + ks * 32, K);
    acc = wmma_bf16(a, b, acc);
  }
  int col = ct + (lane & 15);
  float bi = bias[col];
  int rowoff = (lane < 16) ? 0 : 8;
#pragma unroll
  for (int v = 0; v < 8; ++v) {
    int r = row0 + rowoff + v;
    int rr = transpose ? (((r & 255) << 8) | (r >> 8)) : r;
    pair_out[(size_t)rr * DPAIR + col] = pair_in[(size_t)rr * DPAIR + col] + acc[v] + bi;
  }
}

extern "C" void kernel_launch(void* const* d_in, const int* in_sizes, int n_in,
                              void* d_out, int out_size, void* d_ws, size_t ws_size,
                              hipStream_t stream) {
  (void)in_sizes; (void)n_in; (void)out_size; (void)ws_size;
  const float* pair0 = (const float*)d_in[0];
  float* pair = (float*)d_out;                 // running pair buffer
  char* ws = (char*)d_ws;

  // ---- workspace layout ----
  const size_t OFF_XLN = 0;
  const size_t OFF_W   = OFF_XLN + (size_t)NROWS * DPAIR * sizeof(bf16);
  const size_t OFF_ST  = OFF_W + (1u << 20);
  bf16* xln   = (bf16*)(ws + OFF_XLN);
  // tri-mul region
  bf16* Lmat  = (bf16*)(ws + OFF_ST);
  bf16* Rtm   = Lmat + (size_t)DHID * NROWS;
  bf16* Gate  = Rtm + (size_t)DHID * NROWS;
  float* Tri  = (float*)(Gate + (size_t)NROWS * DPAIR);
  bf16* TriLn = (bf16*)(Tri + (size_t)NROWS * DHID);
  // attention region (aliases tri region)
  bf16* qmat = (bf16*)(ws + OFF_ST);
  bf16* kmat = qmat + (size_t)NROWS * DPAIR;
  bf16* vmat = kmat + (size_t)NROWS * DPAIR;
  bf16* gbuf = vmat + (size_t)NROWS * DPAIR;
  float* bbuf = (float*)(gbuf + (size_t)NROWS * DPAIR);
  bf16* obuf = (bf16*)(bbuf + (size_t)NHEAD * NROWS);
  // ff region (aliases)
  bf16* hbuf = (bf16*)(ws + OFF_ST);

  bf16* wp = (bf16*)(ws + OFF_W);
  size_t wofs = 0;
  auto walloc = [&](size_t n) { bf16* p = wp + wofs; wofs += n; return p; };
  // Cast f32 (K x N) weight to bf16 transposed (ndst x K), zero-padding cols >= nsrc.
  auto castT = [&](int idx, bf16* dst, int K, int nsrc, int ndst) {
    int total = ndst * K;
    castT_kernel<<<(total + 255) / 256, 256, 0, stream>>>((const float*)d_in[idx], dst, K, nsrc, ndst);
  };

  // ---- bf16 transposed weight packing ----
  struct TriW { bf16 *lw, *lgw, *rw, *rgw, *gw, *ow; } triw[2];
  for (int s = 0; s < 2; ++s) {
    int bi = 1 + s * 16;  // ln_g, ln_b, left_w, left_b, lg_w, lg_b, right_w, right_b,
                          // rg_w, rg_b, lno_g, lno_b, out_w, out_b, gate_w, gate_b
    triw[s].lw  = walloc(32 * 128);  castT(bi + 2,  triw[s].lw, 128, 32, 32);
    triw[s].lgw = walloc(32 * 128);  castT(bi + 4,  triw[s].lgw, 128, 32, 32);
    triw[s].rw  = walloc(32 * 128);  castT(bi + 6,  triw[s].rw, 128, 32, 32);
    triw[s].rgw = walloc(32 * 128);  castT(bi + 8,  triw[s].rgw, 128, 32, 32);
    triw[s].ow  = walloc(128 * 32);  castT(bi + 12, triw[s].ow, 32, 128, 128);
    triw[s].gw  = walloc(128 * 128); castT(bi + 14, triw[s].gw, 128, 128, 128);
  }
  struct AttnW { bf16 *qw, *kw, *vw, *gw, *bw, *ow; } attw[2];
  for (int s = 0; s < 2; ++s) {
    int bi = 33 + s * 10;  // ln_g, ln_b, q_w, k_w, v_w, b_w, g_w, g_b, o_w, o_b
    attw[s].qw = walloc(128 * 128); castT(bi + 2, attw[s].qw, 128, 128, 128);
    attw[s].kw = walloc(128 * 128); castT(bi + 3, attw[s].kw, 128, 128, 128);
    attw[s].vw = walloc(128 * 128); castT(bi + 4, attw[s].vw, 128, 128, 128);
    attw[s].bw = walloc(16 * 128);  castT(bi + 5, attw[s].bw, 128, 4, 16);  // zero-padded to 16 rows
    attw[s].gw = walloc(128 * 128); castT(bi + 6, attw[s].gw, 128, 128, 128);
    attw[s].ow = walloc(128 * 128); castT(bi + 8, attw[s].ow, 128, 128, 128);
  }
  bf16* w1 = walloc(256 * 128); castT(55, w1, 128, 256, 256);
  bf16* w2 = walloc(128 * 256); castT(57, w2, 256, 128, 128);

  // ---- triangle multiplicative updates (outgoing, then incoming) ----
  for (int s = 0; s < 2; ++s) {
    int bi = 1 + s * 16;
    const float* psrc = (s == 0) ? pair0 : pair;
    ln128_kernel<<<8192, 256, 0, stream>>>(psrc, (const float*)d_in[bi], (const float*)d_in[bi + 1], xln, 0);
    trimul_proj_kernel<<<dim3(1024, 12), 128, 0, stream>>>(
        xln, triw[s].lw, (const float*)d_in[bi + 3], triw[s].lgw, (const float*)d_in[bi + 5],
        triw[s].rw, (const float*)d_in[bi + 7], triw[s].rgw, (const float*)d_in[bi + 9],
        triw[s].gw, (const float*)d_in[bi + 15], Lmat, Rtm, Gate, (s == 0) ? 1 : 0);
    tri_gemm_kernel<<<256, 256, 0, stream>>>(Lmat, Rtm, Tri);
    ln32_kernel<<<8192, 256, 0, stream>>>(Tri, (const float*)d_in[bi + 10], (const float*)d_in[bi + 11], TriLn);
    trimul_out_kernel<<<dim3(1024, 8), 128, 0, stream>>>(
        TriLn, triw[s].ow, (const float*)d_in[bi + 13], Gate, psrc, pair);
  }

  // ---- triangle attentions (start-node, then end-node via transposed indexing) ----
  for (int s = 0; s < 2; ++s) {
    int bi = 33 + s * 10;
    int tp = s;  // end-node: operate on transposed pair
    ln128_kernel<<<8192, 256, 0, stream>>>(pair, (const float*)d_in[bi], (const float*)d_in[bi + 1], xln, tp);
    attn_proj_kernel<<<dim3(1024, 33), 128, 0, stream>>>(
        xln, attw[s].qw, attw[s].kw, attw[s].vw, attw[s].gw, (const float*)d_in[bi + 7],
        attw[s].bw, qmat, kmat, vmat, gbuf, bbuf);
    attn_flash_kernel<<<dim3(1024, 4), 128, 0, stream>>>(qmat, kmat, vmat, bbuf, gbuf, obuf);
    gemm_resid_kernel<<<dim3(1024, 8), 128, 0, stream>>>(
        obuf, 128, attw[s].ow, (const float*)d_in[bi + 9], pair, pair, tp);
  }

  // ---- feed-forward transition ----
  {
    int bi = 53;
    ln128_kernel<<<8192, 256, 0, stream>>>(pair, (const float*)d_in[bi], (const float*)d_in[bi + 1], xln, 0);
    ff1_kernel<<<dim3(1024, 16), 128, 0, stream>>>(xln, w1, (const float*)d_in[bi + 3], hbuf);
    gemm_resid_kernel<<<dim3(1024, 8), 128, 0, stream>>>(
        hbuf, 256, w2, (const float*)d_in[bi + 5], pair, pair, 0);
  }
}
